// WindowAttention3D_3178275799715
// MI455X (gfx1250) — compile-verified, matrix operand order checked by reading, not run on hardware
//
#include <hip/hip_runtime.h>

typedef __attribute__((ext_vector_type(16))) _Float16 v16h;
typedef __attribute__((ext_vector_type(8)))  _Float16 v8h;
typedef __attribute__((ext_vector_type(8)))  float    v8f;
typedef __attribute__((ext_vector_type(4)))  float    v4f;
typedef __attribute__((ext_vector_type(4)))  unsigned int u32x4;
typedef __attribute__((ext_vector_type(8)))  unsigned int u32x8;

#define HEADS 8
#define HD    32
#define NTOK  98
#define NP    112          // 7*16 padded token count
#define KP    128          // K padded to multiple of 32 for P@V
#define DIM   256
#define BATCH 2048
#define NWIN  256
#define MROWS (BATCH * NTOK)   // 200704
#define MT    (MROWS / 16)     // 12544 (exact)
#define QK_SCALE 0.17677669529663687f  // 32^-0.5

// ---- TDM: 2D tensor_load_to_lds (fp16 elements) --------------------------
// D# per CDNA5 ISA 8.3/8.4. group0: count=1, lds_addr, global_addr, type=2.
// group1: data_size=1(2B), tensor_dim0/1, tile_dim0/1, tensor_dim0_stride.
// OOB reads (tile > tensor extent) return zero -> free zero padding.
__device__ __forceinline__ void tdm_load_2d_f16(unsigned lds_addr, const void* gptr,
                                                unsigned tensor_d0, unsigned tensor_d1,
                                                unsigned tile_d0, unsigned tile_d1,
                                                unsigned stride_d0) {
    unsigned long long ga = (unsigned long long)gptr;
    u32x4 g0;
    g0[0] = 1u;                                               // count=1, user D#
    g0[1] = lds_addr;                                         // LDS byte offset
    g0[2] = (unsigned)ga;                                     // global_addr[31:0]
    g0[3] = (unsigned)((ga >> 32) & 0x1FFFFFFu) | (2u << 30); // addr[56:32] | type=2
    u32x8 g1;
    g1[0] = (1u << 16);                                       // data_size=2B, mask=0
    g1[1] = (tensor_d0 & 0xFFFFu) << 16;                      // tensor_dim0[15:0]
    g1[2] = (tensor_d0 >> 16) | ((tensor_d1 & 0xFFFFu) << 16);
    g1[3] = (tensor_d1 >> 16) | (tile_d0 << 16);              // tile_dim0
    g1[4] = tile_d1;                                          // tile_dim1, tile_dim2=0
    g1[5] = stride_d0;                                        // dim0_stride[31:0]
    g1[6] = 0u;
    g1[7] = 0u;
    u32x4 gz = {0u, 0u, 0u, 0u};
    asm volatile("tensor_load_to_lds %0, %1, %2, %3"
                 :: "s"(g0), "s"(g1), "s"(gz), "s"(gz)
                 : "memory");
}

__device__ __forceinline__ unsigned lds_off(const void* p) {
    return (unsigned)(unsigned long long)p;   // LDS generic addr: offset in [31:0]
}

// ---- fragment loaders ----------------------------------------------------
// A/B lane layout for v_wmma_f32_16x16x32_f16 (wave32):
//   lane l: row/col = l & 15, K-chunks of 8 halves at k0=(l>>4)*8 and k0+16.
__device__ __forceinline__ v16h frag_from_f32(const float* __restrict__ row, int koff) {
    v4f a0 = *(const v4f*)(row + koff);
    v4f a1 = *(const v4f*)(row + koff + 4);
    v4f b0 = *(const v4f*)(row + koff + 16);
    v4f b1 = *(const v4f*)(row + koff + 20);
    v16h r;
#pragma unroll
    for (int i = 0; i < 4; ++i) {
        r[i]      = (_Float16)a0[i];
        r[i + 4]  = (_Float16)a1[i];
        r[i + 8]  = (_Float16)b0[i];
        r[i + 12] = (_Float16)b1[i];
    }
    return r;
}

__device__ __forceinline__ v16h frag_from_f16(const _Float16* row, int koff) {
    v8h lo = *(const v8h*)(row + koff);
    v8h hi = *(const v8h*)(row + koff + 16);
    v16h r;
#pragma unroll
    for (int i = 0; i < 8; ++i) { r[i] = lo[i]; r[i + 8] = hi[i]; }
    return r;
}

// ---- kernel 0: one-time fp32 -> fp16 weight conversion -------------------
__global__ void cvt_weights_kernel(const float* __restrict__ qkv_w,
                                   const float* __restrict__ proj_w,
                                   _Float16* __restrict__ w16,
                                   _Float16* __restrict__ p16) {
    const int i = blockIdx.x * blockDim.x + threadIdx.x;
    if (i < 3 * DIM * DIM) w16[i] = (_Float16)qkv_w[i];
    if (i < DIM * DIM)     p16[i] = (_Float16)proj_w[i];
}

// ---- kernel 1: qkv = x @ qkv_w^T, TDM-staged B, scatter to fp16 layouts --
__global__ __launch_bounds__(256) void qkv_kernel(const float* __restrict__ x,
                                                  const _Float16* __restrict__ w16,
                                                  _Float16* __restrict__ qw,
                                                  _Float16* __restrict__ kw,
                                                  _Float16* __restrict__ vw) {
    __shared__ _Float16 Bs[2][16][DIM];            // 16 KB double buffer
    const int lane = threadIdx.x & 31;
    const int wave = threadIdx.x >> 5;
    const int mt   = blockIdx.x * 8 + wave;        // 12544 m-tiles total
    const int m0   = mt * 16;
    const int kb   = (lane >> 4) * 8;
    const float* aptr = x + (size_t)(m0 + (lane & 15)) * DIM;

    v16h afrag[8];
#pragma unroll
    for (int ks = 0; ks < 8; ++ks) afrag[ks] = frag_from_f32(aptr, kb + ks * 32);

    const unsigned ldsb0 = lds_off(&Bs[0][0][0]);
    const unsigned ldsb1 = lds_off(&Bs[1][0][0]);
    if (wave == 0) {
        tdm_load_2d_f16(ldsb0, w16, DIM, 16, DIM, 16, DIM);
        __builtin_amdgcn_s_wait_tensorcnt(0);
    }
    __syncthreads();

    for (int nt = 0; nt < 48; ++nt) {
        const int cur = nt & 1;
        const bool pf = (nt + 1 < 48);
        if (wave == 0 && pf)                       // prefetch next tile (other buf)
            tdm_load_2d_f16(cur ? ldsb0 : ldsb1, w16 + (size_t)(nt + 1) * 16 * DIM,
                            DIM, 16, DIM, 16, DIM);

        const _Float16* brow = &Bs[cur][lane & 15][0];
        v16h bfrag[8];                             // preload: 16x ds_load_b128
#pragma unroll
        for (int ks = 0; ks < 8; ++ks) bfrag[ks] = frag_from_f16(brow, kb + ks * 32);

        v8f acc = {};                              // then 8 back-to-back WMMAs
#pragma unroll
        for (int ks = 0; ks < 8; ++ks)
            acc = __builtin_amdgcn_wmma_f32_16x16x32_f16(false, afrag[ks], false, bfrag[ks],
                                                         (short)0, acc, false, false);

        // epilogue: q/k/v branch is per-lane invariant across the 8 rows
        const int col   = nt * 16 + (lane & 15);
        const int rbase = m0 + 8 * (lane >> 4);
        if (col < DIM) {                                   // Q (pre-scaled)
            const int h = col >> 5, d = col & 31;
#pragma unroll
            for (int r = 0; r < 8; ++r) {
                const int row = rbase + r;
                const int b = row / NTOK, n = row - b * NTOK;
                qw[(((size_t)b * HEADS + h) * NTOK + n) * HD + d] =
                    (_Float16)(acc[r] * QK_SCALE);
            }
        } else if (col < 2 * DIM) {                        // K
            const int c = col - DIM, h = c >> 5, d = c & 31;
#pragma unroll
            for (int r = 0; r < 8; ++r) {
                const int row = rbase + r;
                const int b = row / NTOK, n = row - b * NTOK;
                kw[(((size_t)b * HEADS + h) * NTOK + n) * HD + d] = (_Float16)acc[r];
            }
        } else {                                           // V transposed [d][tok]
            const int c = col - 2 * DIM, h = c >> 5, d = c & 31;
#pragma unroll
            for (int r = 0; r < 8; ++r) {
                const int row = rbase + r;
                const int b = row / NTOK, n = row - b * NTOK;
                vw[(((size_t)b * HEADS + h) * HD + d) * NTOK + n] = (_Float16)acc[r];
            }
        }
        if (wave == 0 && pf) __builtin_amdgcn_s_wait_tensorcnt(0);
        __syncthreads();
    }
}

// ---- kernel 2: fused attention per (window, head) ------------------------
__global__ __launch_bounds__(128) void attn_kernel(const _Float16* __restrict__ qw,
                                                   const _Float16* __restrict__ kw,
                                                   const _Float16* __restrict__ vw,
                                                   const float* __restrict__ mask,
                                                   const float* __restrict__ pbt,
                                                   const int* __restrict__ rpi,
                                                   _Float16* __restrict__ aout) {
    __shared__ _Float16 Qs[NP][HD];                // 7168 B (rows 98..111 TDM-zeroed)
    __shared__ _Float16 Ks[NP][HD];                // 7168 B
    __shared__ _Float16 Vs[HD][KP];                // 8192 B (cols 98..127 TDM-zeroed)
    __shared__ _Float16 P[NP][KP];                 // 28672 B: logits, then probs

    const int lane = threadIdx.x & 31;
    const int wave = threadIdx.x >> 5;
    const int bh   = blockIdx.x;
    const int b    = bh >> 3;
    const int h    = bh & 7;
    const int w    = b & (NWIN - 1);
    const int kb   = (lane >> 4) * 8;

    // ---- phase 0: TDM the whole (b,h) problem into LDS, OOB zero-padded
    if (wave == 0) {
        tdm_load_2d_f16(lds_off(&Qs[0][0]), qw + (size_t)bh * NTOK * HD,
                        HD, NTOK, HD, NP, HD);
        tdm_load_2d_f16(lds_off(&Ks[0][0]), kw + (size_t)bh * NTOK * HD,
                        HD, NTOK, HD, NP, HD);
        tdm_load_2d_f16(lds_off(&Vs[0][0]), vw + (size_t)bh * HD * NTOK,
                        NTOK, HD, KP, HD, NTOK);
        __builtin_amdgcn_s_wait_tensorcnt(0);
    }
    __syncthreads();

    // ---- phase 1: S = q @ k^T + bias + mask (one WMMA per 16x16 tile, K=32)
    for (int t = wave; t < 49; t += 4) {
        const int mt = t / 7, nt = t - mt * 7;
        const int m0 = mt * 16, n0 = nt * 16;
        v16h a  = frag_from_f16(&Qs[m0 + (lane & 15)][0], kb);
        v16h bf = frag_from_f16(&Ks[n0 + (lane & 15)][0], kb);
        v8f acc = {};
        acc = __builtin_amdgcn_wmma_f32_16x16x32_f16(false, a, false, bf,
                                                     (short)0, acc, false, false);
        const int col = n0 + (lane & 15);
#pragma unroll
        for (int r = 0; r < 8; ++r) {
            const int row = m0 + r + 8 * (lane >> 4);
            float val = -3.0e4f;                   // fp16-representable "-inf"
            if (row < NTOK && col < NTOK) {
                const int idx = rpi[row * NTOK + col];
                val = acc[r] + pbt[idx * HEADS + h]
                             + mask[((size_t)w * NTOK + row) * NTOK + col];
            }
            P[row][col] = (_Float16)val;
        }
    }
    __syncthreads();

    // ---- phase 2: per-row softmax (thread t owns row t), zero K-padding
    {
        const int t = threadIdx.x;
        if (t < NP) {
            if (t < NTOK) {
                float mx = -3.4e38f;
                for (int c = 0; c < NTOK; ++c) mx = fmaxf(mx, (float)P[t][c]);
                float sum = 0.f;
                for (int c = 0; c < NTOK; ++c) {
                    float e = __expf((float)P[t][c] - mx);
                    sum += e;
                    P[t][c] = (_Float16)e;
                }
                const float inv = 1.f / sum;
                for (int c = 0; c < NTOK; ++c) P[t][c] = (_Float16)((float)P[t][c] * inv);
                for (int c = NTOK; c < KP; ++c) P[t][c] = (_Float16)0.f;
            } else {
                for (int c = 0; c < KP; ++c) P[t][c] = (_Float16)0.f;
            }
        }
    }
    __syncthreads();

    // ---- phase 3: out = P @ V  (7 m-tiles x 2 n-tiles, K=128 -> 4 WMMAs)
    for (int t = wave; t < 14; t += 4) {
        const int mt = t >> 1, nt = t & 1;
        const int m0 = mt * 16, n0 = nt * 16;
        v16h a[4], bf[4];                          // preload all fragments
#pragma unroll
        for (int ks = 0; ks < 4; ++ks) {
            const int koff = kb + ks * 32;
            a[ks]  = frag_from_f16(&P[m0 + (lane & 15)][0], koff);
            bf[ks] = frag_from_f16(&Vs[n0 + (lane & 15)][0], koff);
        }
        v8f acc = {};
#pragma unroll
        for (int ks = 0; ks < 4; ++ks)
            acc = __builtin_amdgcn_wmma_f32_16x16x32_f16(false, a[ks], false, bf[ks],
                                                         (short)0, acc, false, false);
        const int col = n0 + (lane & 15);
#pragma unroll
        for (int r = 0; r < 8; ++r) {
            const int row = m0 + r + 8 * (lane >> 4);
            if (row < NTOK)
                aout[((size_t)b * NTOK + row) * DIM + h * HD + col] = (_Float16)acc[r];
        }
    }
}

// ---- kernel 3: out = attn @ proj_w^T + proj_b, TDM-staged B --------------
__global__ __launch_bounds__(256) void proj_kernel(const _Float16* __restrict__ a,
                                                   const _Float16* __restrict__ p16,
                                                   const float* __restrict__ pb,
                                                   float* __restrict__ out) {
    __shared__ _Float16 Bs[2][16][DIM];
    const int lane = threadIdx.x & 31;
    const int wave = threadIdx.x >> 5;
    const int mt   = blockIdx.x * 8 + wave;
    const int m0   = mt * 16;
    const int kb   = (lane >> 4) * 8;
    const _Float16* aptr = a + (size_t)(m0 + (lane & 15)) * DIM;

    v16h afrag[8];
#pragma unroll
    for (int ks = 0; ks < 8; ++ks) afrag[ks] = frag_from_f16(aptr, kb + ks * 32);

    const unsigned ldsb0 = lds_off(&Bs[0][0][0]);
    const unsigned ldsb1 = lds_off(&Bs[1][0][0]);
    if (wave == 0) {
        tdm_load_2d_f16(ldsb0, p16, DIM, 16, DIM, 16, DIM);
        __builtin_amdgcn_s_wait_tensorcnt(0);
    }
    __syncthreads();

    for (int nt = 0; nt < 16; ++nt) {
        const int cur = nt & 1;
        const bool pf = (nt + 1 < 16);
        if (wave == 0 && pf)
            tdm_load_2d_f16(cur ? ldsb0 : ldsb1, p16 + (size_t)(nt + 1) * 16 * DIM,
                            DIM, 16, DIM, 16, DIM);

        const _Float16* brow = &Bs[cur][lane & 15][0];
        v16h bfrag[8];
#pragma unroll
        for (int ks = 0; ks < 8; ++ks) bfrag[ks] = frag_from_f16(brow, kb + ks * 32);

        v8f acc = {};
#pragma unroll
        for (int ks = 0; ks < 8; ++ks)
            acc = __builtin_amdgcn_wmma_f32_16x16x32_f16(false, afrag[ks], false, bfrag[ks],
                                                         (short)0, acc, false, false);

        const int col  = nt * 16 + (lane & 15);
        const float bias = pb[col];
#pragma unroll
        for (int r = 0; r < 8; ++r) {
            const int row = m0 + r + 8 * (lane >> 4);
            out[(size_t)row * DIM + col] = acc[r] + bias;
        }
        if (wave == 0 && pf) __builtin_amdgcn_s_wait_tensorcnt(0);
        __syncthreads();
    }
}

// ---- host launcher -------------------------------------------------------
extern "C" void kernel_launch(void* const* d_in, const int* in_sizes, int n_in,
                              void* d_out, int out_size, void* d_ws, size_t ws_size,
                              hipStream_t stream) {
    const float* x      = (const float*)d_in[0];
    const float* mask   = (const float*)d_in[1];
    const float* pbt    = (const float*)d_in[2];
    const float* qkv_w  = (const float*)d_in[3];
    const float* proj_w = (const float*)d_in[4];
    const float* proj_b = (const float*)d_in[5];
    const int*   rpi    = (const int*)d_in[6];
    float* out = (float*)d_out;

    char* ws = (char*)d_ws;
    const size_t SZQ = (size_t)BATCH * HEADS * NTOK * HD * sizeof(_Float16); // 102.8 MB
    _Float16* qw   = (_Float16*)ws;
    _Float16* kw   = (_Float16*)(ws + SZQ);
    _Float16* vw   = (_Float16*)(ws + 2 * SZQ);
    _Float16* aout = (_Float16*)(ws + 3 * SZQ);
    _Float16* w16  = (_Float16*)(ws + 4 * SZQ);
    _Float16* p16  = (_Float16*)(ws + 4 * SZQ + (size_t)3 * DIM * DIM * sizeof(_Float16));

    cvt_weights_kernel<<<(3 * DIM * DIM) / 256, 256, 0, stream>>>(qkv_w, proj_w, w16, p16);
    qkv_kernel<<<MT / 8, 256, 0, stream>>>(x, w16, qw, kw, vw);
    attn_kernel<<<BATCH * HEADS, 128, 0, stream>>>(qw, kw, vw, mask, pbt, rpi, aout);
    proj_kernel<<<MT / 8, 256, 0, stream>>>(aout, p16, proj_b, out);
}
